// MoE_46334107189528
// MI455X (gfx1250) — matrix operand order repaired and checked by reading, not compiled, not run on hardware
//
#include <hip/hip_runtime.h>
#include <hip/hip_bf16.h>
#include <math.h>

// ---------------- problem constants ----------------
#define DIMX 1024
#define HID  1024
#define NE   8
#define NTOK 2048          // 2*1024 tokens
#define TOPK 2
#define CAP_E 2048         // max rows per expert (each token appears <=1x per expert)
#define ROUTED_CAP 4224    // 4096 routed rows + 8*16 padding headroom (16-aligned)
#define SHARED_START ROUTED_CAP
#define TOTAL_ROWS (ROUTED_CAP + NTOK)   // 6272
#define NTILES (TOTAL_ROWS / 16)         // 392

// ---------------- ws layout (bytes, 256-aligned regions) ----------------
#define OFF_COUNTS 0
#define OFF_OFF    256
#define OFF_TILEE  512
#define OFF_INV    4096
#define OFF_BTOK   20480
#define OFF_BSCL   86016
#define OFF_XB     151552        // bf16 [6272*1024]
#define OFF_H      12996608      // bf16 [6272*1024]
#define OFF_O      25841664      // f32  [6272*1024]  -> total ~49.2 MB

typedef __attribute__((ext_vector_type(16))) __bf16 v16bf;
typedef __attribute__((ext_vector_type(4)))  __bf16 v4bf;
typedef __attribute__((ext_vector_type(8)))  float  v8f;

// CDNA5 async global->LDS copies (ASYNCcnt-tracked, no VGPR staging).
// SADDR form: vdst = per-lane LDS byte address, voffset = 32-bit byte offset, s[base].
#define ASYNC_B128(ldsoff, base, byteoff)                                  \
  asm volatile("global_load_async_to_lds_b128 %0, %1, %2"                  \
               :: "v"(ldsoff), "v"(byteoff), "s"(base) : "memory")
#define ASYNC_B32(ldsoff, base, byteoff)                                   \
  asm volatile("global_load_async_to_lds_b32 %0, %1, %2"                   \
               :: "v"(ldsoff), "v"(byteoff), "s"(base) : "memory")
#define WAIT_ASYNC(n) asm volatile("s_wait_asynccnt " #n ::: "memory")

__device__ __forceinline__ unsigned lds_addr_of(const void* p) {
  // flat LDS aperture keeps the LDS offset in addr[31:0] (ISA §10.2)
  return (unsigned)(size_t)p;
}

// ---------------- kernel 0: zero expert counters ----------------
__global__ void init_k(int* counts) {
  if (threadIdx.x < NE) counts[threadIdx.x] = 0;
}

// ---------------- kernel 1: router (fp32 logits, softmax, top-2) ----------------
__global__ void router_k(const float* __restrict__ x, const float* __restrict__ gw,
                         int* __restrict__ counts, int* __restrict__ btok,
                         float* __restrict__ bscl) {
  int t = blockIdx.x * blockDim.x + threadIdx.x;
  if (t >= NTOK) return;
  const float* xr = x + (size_t)t * DIMX;
  float lg[NE];
  #pragma unroll
  for (int e = 0; e < NE; ++e) {
    const float* g = gw + e * DIMX;
    float s = 0.f;
    for (int i = 0; i < DIMX; ++i) s = fmaf(xr[i], g[i], s);
    lg[e] = s;
  }
  float mx = lg[0];
  #pragma unroll
  for (int e = 1; e < NE; ++e) mx = fmaxf(mx, lg[e]);
  float p[NE], sum = 0.f;
  #pragma unroll
  for (int e = 0; e < NE; ++e) { p[e] = __expf(lg[e] - mx); sum += p[e]; }
  float inv_s = 1.f / sum;
  #pragma unroll
  for (int e = 0; e < NE; ++e) p[e] *= inv_s;
  int i0 = 0;
  #pragma unroll
  for (int e = 1; e < NE; ++e) if (p[e] > p[i0]) i0 = e;
  int i1 = (i0 == 0) ? 1 : 0;
  #pragma unroll
  for (int e = 0; e < NE; ++e) if (e != i0 && p[e] > p[i1]) i1 = e;
  int   sel[TOPK] = { i0, i1 };
  float scr[TOPK] = { p[i0], p[i1] };
  #pragma unroll
  for (int k = 0; k < TOPK; ++k) {
    int e = sel[k];
    int slot = atomicAdd(&counts[e], 1);
    btok[e * CAP_E + slot] = t * TOPK + k;
    bscl[e * CAP_E + slot] = scr[k];
  }
}

// ---------------- kernel 2: offsets + tile->expert map ----------------
__global__ void setup_k(const int* __restrict__ counts, int* __restrict__ off,
                        int* __restrict__ tile_e) {
  if (threadIdx.x != 0 || blockIdx.x != 0) return;
  int offs[NE + 1];
  int o = 0;
  for (int e = 0; e < NE; ++e) { offs[e] = o; o += (counts[e] + 15) & ~15; }
  offs[NE] = o;
  for (int e = 0; e <= NE; ++e) off[e] = offs[e];
  for (int tile = 0; tile < NTILES; ++tile) {
    int r = tile * 16, te;
    if (r >= SHARED_START)      te = NE;
    else if (r >= offs[NE])     te = -1;
    else {
      te = -1;
      for (int e = 0; e < NE; ++e)
        if (r >= offs[e] && r < offs[e + 1]) te = e;
    }
    tile_e[tile] = te;
  }
}

// ---------------- kernel 3: gather + pre-scale -> bf16 A matrix; inverse map ----------------
__global__ void gather_k(const float* __restrict__ x, const int* __restrict__ counts,
                         const int* __restrict__ off, const int* __restrict__ tile_e,
                         const int* __restrict__ btok, const float* __restrict__ bscl,
                         int* __restrict__ inv, __bf16* __restrict__ Xb) {
  int r = blockIdx.x;
  int e = tile_e[r >> 4];
  int tok = 0; float scale = 0.f;
  if (e == NE) { tok = r - SHARED_START; scale = 1.f; }
  else if (e >= 0) {
    int slot = r - off[e];
    if (slot < counts[e]) {
      int tk = btok[e * CAP_E + slot];
      tok = tk >> 1; scale = bscl[e * CAP_E + slot];
      if (threadIdx.x == 0) inv[tk] = r;
    }
  }
  int c = threadIdx.x * 4;
  float4 v = *(const float4*)(x + (size_t)tok * DIMX + c);
  v4bf o = { (__bf16)(v.x * scale), (__bf16)(v.y * scale),
             (__bf16)(v.z * scale), (__bf16)(v.w * scale) };
  *(v4bf*)(Xb + (size_t)r * DIMX + c) = o;
}

// ---------------- kernel 4: fused SwiGLU GEMM-1  H = silu(X@W1) * (X@W3) ----------------
// block = 256 = 8 waves; tile = 16 rows x 128 cols; double-buffered async LDS staging.
__global__ void __launch_bounds__(256) gemm1_k(const __bf16* __restrict__ Xb,
    const float* __restrict__ w1, const float* __restrict__ w3,
    const float* __restrict__ w1s, const float* __restrict__ w3s,
    const int* __restrict__ tile_e, __bf16* __restrict__ H) {
  __shared__ __align__(16) __bf16 sA [2][16 * 32];
  __shared__ __align__(16) float  sB1[2][32 * 128];
  __shared__ __align__(16) float  sB3[2][32 * 128];
  int tile = blockIdx.y;
  int e = tile_e[tile];
  if (e < 0) return;                                   // block-uniform exit
  const float* W1 = (e == NE) ? w1s : w1 + (size_t)e * DIMX * HID;
  const float* W3 = (e == NE) ? w3s : w3 + (size_t)e * DIMX * HID;
  const int tid  = threadIdx.x;
  const int wave = tid >> 5, lane = tid & 31;
  const int row0 = tile * 16;
  const int nbase = blockIdx.x * 128;
  const int hsel = lane >> 4;
  const int wcol = wave * 16 + (lane & 15);
  const int ncol = nbase + wcol;
  const int sa_m = tid >> 4;
  const int sa_k = (tid & 15) << 1;
  const unsigned aBase  = lds_addr_of(&sA[0][0]);
  const unsigned b1Base = lds_addr_of(&sB1[0][0]);
  const unsigned b3Base = lds_addr_of(&sB3[0][0]);

  auto issue = [&](int buf, int k0) {
    // A tile: one dword per thread (coalesced along k), async to LDS
    unsigned aOff = aBase + (unsigned)buf * (16 * 32 * 2) +
                    (unsigned)(sa_m * 32 + sa_k) * 2;
    unsigned gA = (unsigned)(((row0 + sa_m) * DIMX + k0 + sa_k) * 2);
    ASYNC_B32(aOff, Xb, gA);
    // W1/W3 tiles: 4x b128 per thread each, fully coalesced
    #pragma unroll
    for (int r = 0; r < 4; ++r) {
      int idx = tid + r * 256;
      int kk  = idx >> 5;
      int c4  = (idx & 31) << 2;
      unsigned goff = (unsigned)(((k0 + kk) * HID + nbase + c4) * 4);
      unsigned loff = (unsigned)(idx * 16) + (unsigned)buf * 16384;
      ASYNC_B128(b1Base + loff, W1, goff);
      ASYNC_B128(b3Base + loff, W3, goff);
      if (r == 0 && k0 + 32 < DIMX) {   // warm L2 one more tile ahead
        __builtin_prefetch(W1 + (size_t)(k0 + 32 + kk) * HID + nbase + c4, 0, 1);
        __builtin_prefetch(W3 + (size_t)(k0 + 32 + kk) * HID + nbase + c4, 0, 1);
      }
    }
  };

  v8f acc1 = {}, acc3 = {};
  issue(0, 0);
  for (int k0 = 0; k0 < DIMX; k0 += 32) {
    int buf = (k0 >> 5) & 1;
    if (k0 + 32 < DIMX) {
      issue(buf ^ 1, k0 + 32);     // stage i+1 in flight...
      WAIT_ASYNC(0x9);             // ...only stage i (oldest 9) must be done
    } else {
      WAIT_ASYNC(0x0);
    }
    __syncthreads();               // all waves' async writes visible
    union { uint4 q[2]; v16bf v; } A;
    A.q[0] = *(const uint4*)(&sA[buf][(lane & 15) * 32 + hsel * 8]);
    A.q[1] = *(const uint4*)(&sA[buf][(lane & 15) * 32 + 16 + hsel * 8]);
    v16bf b1, b3;
    #pragma unroll
    for (int j = 0; j < 16; ++j) {
      int kk = (j & 7) + hsel * 8 + (j >> 3) * 16;
      b1[j] = (__bf16)sB1[buf][kk * 128 + wcol];
      b3[j] = (__bf16)sB3[buf][kk * 128 + wcol];
    }
    acc1 = __builtin_amdgcn_wmma_f32_16x16x32_bf16(false, A.v, false, b1,
                                                   (short)0, acc1, false, false);
    acc3 = __builtin_amdgcn_wmma_f32_16x16x32_bf16(false, A.v, false, b3,
                                                   (short)0, acc3, false, false);
    __syncthreads();               // buffer fully consumed before refill
  }
  #pragma unroll
  for (int rr = 0; rr < 8; ++rr) {                     // C/D layout: M = rr + hsel*8
    int mm = row0 + rr + hsel * 8;
    float g = acc1[rr], u = acc3[rr];
    float sig = __builtin_amdgcn_rcpf(1.f + __expf(-g));
    H[(size_t)mm * HID + ncol] = (__bf16)(g * sig * u);
  }
}

// ---------------- kernel 5: GEMM-2  O = H @ W2 ----------------
__global__ void __launch_bounds__(256) gemm2_k(const __bf16* __restrict__ H,
    const float* __restrict__ w2, const float* __restrict__ w2s,
    const int* __restrict__ tile_e, float* __restrict__ O) {
  __shared__ __align__(16) __bf16 sA [2][16 * 32];
  __shared__ __align__(16) float  sB2[2][32 * 128];
  int tile = blockIdx.y;
  int e = tile_e[tile];
  if (e < 0) return;
  const float* W2 = (e == NE) ? w2s : w2 + (size_t)e * HID * DIMX;
  const int tid  = threadIdx.x;
  const int wave = tid >> 5, lane = tid & 31;
  const int row0 = tile * 16;
  const int nbase = blockIdx.x * 128;
  const int hsel = lane >> 4;
  const int wcol = wave * 16 + (lane & 15);
  const int ncol = nbase + wcol;
  const int sa_m = tid >> 4;
  const int sa_k = (tid & 15) << 1;
  const unsigned aBase  = lds_addr_of(&sA[0][0]);
  const unsigned b2Base = lds_addr_of(&sB2[0][0]);

  auto issue = [&](int buf, int k0) {
    unsigned aOff = aBase + (unsigned)buf * (16 * 32 * 2) +
                    (unsigned)(sa_m * 32 + sa_k) * 2;
    unsigned gA = (unsigned)(((row0 + sa_m) * HID + k0 + sa_k) * 2);
    ASYNC_B32(aOff, H, gA);
    #pragma unroll
    for (int r = 0; r < 4; ++r) {
      int idx = tid + r * 256;
      int kk  = idx >> 5;
      int c4  = (idx & 31) << 2;
      unsigned goff = (unsigned)(((k0 + kk) * DIMX + nbase + c4) * 4);
      unsigned loff = (unsigned)(idx * 16) + (unsigned)buf * 16384;
      ASYNC_B128(b2Base + loff, W2, goff);
      if (r == 0 && k0 + 32 < HID)
        __builtin_prefetch(W2 + (size_t)(k0 + 32 + kk) * DIMX + nbase + c4, 0, 1);
    }
  };

  v8f acc = {};
  issue(0, 0);
  for (int k0 = 0; k0 < HID; k0 += 32) {
    int buf = (k0 >> 5) & 1;
    if (k0 + 32 < HID) {
      issue(buf ^ 1, k0 + 32);
      WAIT_ASYNC(0x5);
    } else {
      WAIT_ASYNC(0x0);
    }
    __syncthreads();
    union { uint4 q[2]; v16bf v; } A;
    A.q[0] = *(const uint4*)(&sA[buf][(lane & 15) * 32 + hsel * 8]);
    A.q[1] = *(const uint4*)(&sA[buf][(lane & 15) * 32 + 16 + hsel * 8]);
    v16bf b;
    #pragma unroll
    for (int j = 0; j < 16; ++j) {
      int kk = (j & 7) + hsel * 8 + (j >> 3) * 16;
      b[j] = (__bf16)sB2[buf][kk * 128 + wcol];
    }
    acc = __builtin_amdgcn_wmma_f32_16x16x32_bf16(false, A.v, false, b,
                                                  (short)0, acc, false, false);
    __syncthreads();
  }
  #pragma unroll
  for (int rr = 0; rr < 8; ++rr) {
    int mm = row0 + rr + hsel * 8;
    O[(size_t)mm * DIMX + ncol] = acc[rr];
  }
}

// ---------------- kernel 6: combine (no atomics): out = shared + routed0 + routed1 ----
__global__ void combine_k(const float* __restrict__ O, const int* __restrict__ inv,
                          float* __restrict__ out) {
  int t = blockIdx.x;
  int c = threadIdx.x * 4;
  int r1 = inv[2 * t], r2 = inv[2 * t + 1];
  float4 a = *(const float4*)(O + (size_t)(SHARED_START + t) * DIMX + c);
  float4 b = *(const float4*)(O + (size_t)r1 * DIMX + c);
  float4 d = *(const float4*)(O + (size_t)r2 * DIMX + c);
  float4 s;
  s.x = a.x + b.x + d.x; s.y = a.y + b.y + d.y;
  s.z = a.z + b.z + d.z; s.w = a.w + b.w + d.w;
  *(float4*)(out + (size_t)t * DIMX + c) = s;
}

extern "C" void kernel_launch(void* const* d_in, const int* in_sizes, int n_in,
                              void* d_out, int out_size, void* d_ws, size_t ws_size,
                              hipStream_t stream) {
  const float* x   = (const float*)d_in[0];
  const float* gw  = (const float*)d_in[1];
  const float* w1  = (const float*)d_in[2];
  const float* w2  = (const float*)d_in[3];
  const float* w3  = (const float*)d_in[4];
  const float* w1s = (const float*)d_in[5];
  const float* w2s = (const float*)d_in[6];
  const float* w3s = (const float*)d_in[7];
  float* out = (float*)d_out;

  char* ws = (char*)d_ws;
  int*    counts = (int*)(ws + OFF_COUNTS);
  int*    off    = (int*)(ws + OFF_OFF);
  int*    tile_e = (int*)(ws + OFF_TILEE);
  int*    inv    = (int*)(ws + OFF_INV);
  int*    btok   = (int*)(ws + OFF_BTOK);
  float*  bscl   = (float*)(ws + OFF_BSCL);
  __bf16* Xb     = (__bf16*)(ws + OFF_XB);
  __bf16* H      = (__bf16*)(ws + OFF_H);
  float*  O      = (float*)(ws + OFF_O);

  init_k<<<1, 32, 0, stream>>>(counts);
  router_k<<<NTOK / 256, 256, 0, stream>>>(x, gw, counts, btok, bscl);
  setup_k<<<1, 1, 0, stream>>>(counts, off, tile_e);
  gather_k<<<TOTAL_ROWS, 256, 0, stream>>>(x, counts, off, tile_e, btok, bscl, inv, Xb);
  dim3 gg(HID / (16 * 8), NTILES);   // (8, 392)
  gemm1_k<<<gg, 256, 0, stream>>>(Xb, w1, w3, w1s, w3s, tile_e, H);
  gemm2_k<<<gg, 256, 0, stream>>>(H, w2, w2s, tile_e, O);
  combine_k<<<NTOK, 256, 0, stream>>>(O, inv, out);
}
